// LLCMBackbone_67688684585102
// MI455X (gfx1250) — matrix operand order repaired and checked by reading, not compiled, not run on hardware
//
#include <hip/hip_runtime.h>
#include <math.h>

#define TSEQ   2048
#define BBATCH 4
#define DMODEL 128
#define NHEADS 4
#define HDIM   32
#define THEAD  8
#define DFF    512
#define NTOK   (BBATCH * TSEQ)   // 8192

typedef __attribute__((ext_vector_type(16))) __bf16 v16bf;
typedef __attribute__((ext_vector_type(8)))  __bf16 v8bf;
typedef __attribute__((ext_vector_type(8)))  float  v8f;

union FragBF { v16bf v; float4 f4[2]; };

__device__ __forceinline__ v16bf ld_frag(const __bf16* p0, const __bf16* p1) {
  FragBF f;
  f.f4[0] = *(const float4*)p0;
  f.f4[1] = *(const float4*)p1;
  return f.v;
}

__device__ __forceinline__ v8f vzero8() {
  v8f z;
#pragma unroll
  for (int r = 0; r < 8; ++r) z[r] = 0.0f;
  return z;
}

__device__ __forceinline__ v8bf pack8(float4 a, float4 b) {
  v8bf p;
  p[0] = (__bf16)a.x; p[1] = (__bf16)a.y; p[2] = (__bf16)a.z; p[3] = (__bf16)a.w;
  p[4] = (__bf16)b.x; p[5] = (__bf16)b.y; p[6] = (__bf16)b.z; p[7] = (__bf16)b.w;
  return p;
}

__device__ __forceinline__ v8bf pack8s(float4 a, float4 b, float s) {
  v8bf p;
  p[0] = (__bf16)(a.x * s); p[1] = (__bf16)(a.y * s);
  p[2] = (__bf16)(a.z * s); p[3] = (__bf16)(a.w * s);
  p[4] = (__bf16)(b.x * s); p[5] = (__bf16)(b.y * s);
  p[6] = (__bf16)(b.z * s); p[7] = (__bf16)(b.w * s);
  return p;
}

__device__ __forceinline__ v8f wmma_bf16(v16bf a, v16bf b, v8f c) {
  return __builtin_amdgcn_wmma_f32_16x16x32_bf16(false, a, false, b, (short)0, c,
                                                 false, false);
}

// ---------------------------------------------------------------------------
// Embed: h[row, d] = sum_k x[row, k] * W[k, d] + b[d]   (K = 6, VALU kernel)
// ---------------------------------------------------------------------------
__launch_bounds__(256)
__global__ void embed_kernel(const float* __restrict__ x, const float* __restrict__ W,
                             const float* __restrict__ bvec, float* __restrict__ h) {
  const int i = blockIdx.x * 256 + threadIdx.x;   // over NTOK*DMODEL
  const int row = i >> 7;
  const int d   = i & 127;
  float acc = bvec[d];
  const float* xr = x + (size_t)row * 6;
#pragma unroll
  for (int k = 0; k < 6; ++k) acc += xr[k] * W[k * DMODEL + d];
  h[i] = acc;
}

// ---------------------------------------------------------------------------
// Generic bf16-WMMA GEMM: C(MxN) = A(MxK) @ W(KxN) + bias [, GELU][, +residual]
// BM=128, BN=64, BK=32; 256 threads = 8 waves in a 4x2 (M x N) arrangement,
// each wave owns a 32x32 output tile = 2x2 v_wmma_f32_16x16x32_bf16 tiles.
// All LDS staging stores are packed ds_store_b128 (8 x bf16).
// scatter=1 reshapes output rows (b*T+t, h*32+d) -> (b,h,t,d) for attention.
// ---------------------------------------------------------------------------
__launch_bounds__(256)
__global__ void gemm_kernel(const float* __restrict__ A, const float* __restrict__ W,
                            const float* __restrict__ bias, const float* __restrict__ residual,
                            float* __restrict__ C, int M, int N, int K, int act, int scatter) {
  __shared__ __align__(16) __bf16 As[128][48];   // [m][k], 96B rows (16B multiple)
  __shared__ __align__(16) __bf16 Bs[64][48];    // [n][k]
  const int m0 = blockIdx.x * 128;
  const int n0 = blockIdx.y * 64;
  const int tid  = threadIdx.x;
  const int lane = tid & 31, wave = tid >> 5;
  const int wm = wave & 3, wn = wave >> 2;
  const int half = lane >> 4, l16 = lane & 15;

  v8f acc[2][2];
#pragma unroll
  for (int i = 0; i < 2; ++i)
#pragma unroll
    for (int j = 0; j < 2; ++j) acc[i][j] = vzero8();

  for (int k0 = 0; k0 < K; k0 += 32) {
    __syncthreads();
    {
      const int r  = tid >> 1;
      const int c0 = (tid & 1) * 16;
      const float* base = A + (size_t)(m0 + r) * K + k0 + c0;
      if (k0 + 32 < K) __builtin_prefetch(base + 32, 0, 1);   // global_prefetch_b8
      const float4* src = (const float4*)base;
      float4 t0 = src[0], t1 = src[1], t2 = src[2], t3 = src[3];
      *(v8bf*)&As[r][c0]     = pack8(t0, t1);
      *(v8bf*)&As[r][c0 + 8] = pack8(t2, t3);
    }
    {
      // thread owns one (n, 8-wide k strip): 8 strided b32 loads (L2-resident
      // weights), one packed ds_store_b128.
      const int n  = tid >> 2;          // 0..63
      const int kg = (tid & 3) * 8;     // 0,8,16,24
      const float* wp = W + (size_t)(k0 + kg) * N + n0 + n;
      float4 a, b;
      a.x = wp[0];
      a.y = wp[(size_t)N];
      a.z = wp[(size_t)2 * N];
      a.w = wp[(size_t)3 * N];
      b.x = wp[(size_t)4 * N];
      b.y = wp[(size_t)5 * N];
      b.z = wp[(size_t)6 * N];
      b.w = wp[(size_t)7 * N];
      *(v8bf*)&Bs[n][kg] = pack8(a, b);
    }
    __syncthreads();

    v16bf af[2], bfr[2];
#pragma unroll
    for (int mt = 0; mt < 2; ++mt) {
      const int row = wm * 32 + mt * 16 + l16;
      af[mt] = ld_frag(&As[row][half * 8], &As[row][16 + half * 8]);
    }
#pragma unroll
    for (int nt = 0; nt < 2; ++nt) {
      const int col = wn * 32 + nt * 16 + l16;
      bfr[nt] = ld_frag(&Bs[col][half * 8], &Bs[col][16 + half * 8]);
    }
#pragma unroll
    for (int mt = 0; mt < 2; ++mt)
#pragma unroll
      for (int nt = 0; nt < 2; ++nt)
        acc[mt][nt] = wmma_bf16(af[mt], bfr[nt], acc[mt][nt]);
  }

#pragma unroll
  for (int mt = 0; mt < 2; ++mt)
#pragma unroll
    for (int nt = 0; nt < 2; ++nt)
#pragma unroll
      for (int r = 0; r < 8; ++r) {
        const int m = m0 + wm * 32 + mt * 16 + r + 8 * half;
        const int n = n0 + wn * 32 + nt * 16 + l16;
        float v = acc[mt][nt][r] + bias[n];
        if (act) v = 0.5f * v * (1.0f + erff(v * 0.70710678118654752f)); // exact GELU
        if (residual) v += residual[(size_t)m * N + n];
        size_t idx;
        if (scatter) {
          const int b = m >> 11, t = m & (TSEQ - 1);
          const int hh = n >> 5, d = n & 31;
          idx = ((size_t)(b * NHEADS + hh) * TSEQ + t) * HDIM + d;
        } else {
          idx = (size_t)m * N + n;
        }
        C[idx] = v;
      }
}

// ---------------------------------------------------------------------------
// Causal flash attention with Lorentz-signed scores.
// Q,K,V in (B*H, T, 32). score = (q . (s (.) k)) / sqrt(32), s = +1 (d<8), -1 else.
// One workgroup per (128-query block, bh); each of the 8 waves owns 16 q-rows.
// ---------------------------------------------------------------------------
__launch_bounds__(256)
__global__ void attn_kernel(const float* __restrict__ Q, const float* __restrict__ Kb,
                            const float* __restrict__ Vb, float* __restrict__ O) {
  __shared__ __align__(16) __bf16 Qs[128][48];       // [qrow][d] (scaled)
  __shared__ __align__(16) __bf16 Ks[128][48];       // [key][d]  (sign folded)
  __shared__ __align__(16) __bf16 Vs[32][136];       // [d][key]  (transposed)
  __shared__ __align__(16) __bf16 Ps[8][16][136];    // per-wave P tile [m][key]

  const int qb = blockIdx.x;                 // 0..15
  const int bh = blockIdx.y;                 // 0..15
  const int b = bh >> 2, hh = bh & 3;
  const int tid  = threadIdx.x;
  const int lane = tid & 31, wave = tid >> 5;
  const int half = lane >> 4, l16 = lane & 15;

  const float scale = 0.17677669529663687f;  // 1/sqrt(32)
  {
    const int r  = tid >> 1;
    const int c0 = (tid & 1) * 16;
    const float4* src = (const float4*)(Q + ((size_t)bh * TSEQ + qb * 128 + r) * HDIM + c0);
    float4 t0 = src[0], t1 = src[1], t2 = src[2], t3 = src[3];
    *(v8bf*)&Qs[r][c0]     = pack8s(t0, t1, scale);
    *(v8bf*)&Qs[r][c0 + 8] = pack8s(t2, t3, scale);
  }
  __syncthreads();

  const v16bf qf = ld_frag(&Qs[wave * 16 + l16][half * 8],
                           &Qs[wave * 16 + l16][16 + half * 8]);

  v8f o0 = vzero8(), o1 = vzero8();
  float mrun[8], lrun[8];
#pragma unroll
  for (int r = 0; r < 8; ++r) { mrun[r] = -1e30f; lrun[r] = 0.0f; }
  const int qrow0 = qb * 128 + wave * 16 + 8 * half;   // row for vgpr r is qrow0+r

  for (int kb = 0; kb <= qb; ++kb) {
    __syncthreads();   // protect Ks/Vs from previous iteration readers
    {
      const int r  = tid >> 1;
      const int c0 = (tid & 1) * 16;
      const float4* ks = (const float4*)(Kb + ((size_t)bh * TSEQ + kb * 128 + r) * HDIM + c0);
      const float4* vs = (const float4*)(Vb + ((size_t)bh * TSEQ + kb * 128 + r) * HDIM + c0);
      float4 k0 = ks[0], k1 = ks[1], k2 = ks[2], k3 = ks[3];
      float4 v0 = vs[0], v1 = vs[1], v2 = vs[2], v3 = vs[3];
      // Lorentz sign: dims [0,8) keep sign, dims [8,32) negate. For the packed
      // group starting at c0: elems c0..c0+7 flip iff c0>=8; c0+8..c0+15 always flip.
      const float sA = (c0 >= THEAD) ? -1.0f : 1.0f;
      *(v8bf*)&Ks[r][c0]     = pack8s(k0, k1, sA);
      *(v8bf*)&Ks[r][c0 + 8] = pack8s(k2, k3, -1.0f);
      // V transposed into [d][key] (scattered b16 stores, needed for B-layout)
      const float vv[16] = { v0.x, v0.y, v0.z, v0.w, v1.x, v1.y, v1.z, v1.w,
                             v2.x, v2.y, v2.z, v2.w, v3.x, v3.y, v3.z, v3.w };
#pragma unroll
      for (int j = 0; j < 16; ++j) Vs[c0 + j][r] = (__bf16)vv[j];
    }
    __syncthreads();

    // S = Q * K'^T : 8 n-tiles, K-dim = 32 -> one WMMA each
    v8f s[8];
#pragma unroll
    for (int nt = 0; nt < 8; ++nt) {
      const int col = nt * 16 + l16;
      v16bf kf = ld_frag(&Ks[col][half * 8], &Ks[col][16 + half * 8]);
      s[nt] = wmma_bf16(qf, kf, vzero8());
    }

    if (kb == qb) {   // causal mask on diagonal block
#pragma unroll
      for (int nt = 0; nt < 8; ++nt) {
        const int kcol = kb * 128 + nt * 16 + l16;
#pragma unroll
        for (int r = 0; r < 8; ++r)
          if (kcol > qrow0 + r) s[nt][r] = -1.0e9f;
      }
    }

    // online softmax (rows live in vgpr r across the 16-lane half)
    float mx[8];
#pragma unroll
    for (int r = 0; r < 8; ++r) {
      float m = s[0][r];
#pragma unroll
      for (int nt = 1; nt < 8; ++nt) m = fmaxf(m, s[nt][r]);
      mx[r] = m;
    }
#pragma unroll
    for (int off = 8; off >= 1; off >>= 1)
#pragma unroll
      for (int r = 0; r < 8; ++r) mx[r] = fmaxf(mx[r], __shfl_xor(mx[r], off, 32));

    float alpha[8], rs[8];
#pragma unroll
    for (int r = 0; r < 8; ++r) {
      const float nm = fmaxf(mrun[r], mx[r]);
      alpha[r] = __expf(mrun[r] - nm);
      mrun[r] = nm;
      rs[r] = 0.0f;
    }
#pragma unroll
    for (int nt = 0; nt < 8; ++nt)
#pragma unroll
      for (int r = 0; r < 8; ++r) {
        const float p = __expf(s[nt][r] - mrun[r]);
        s[nt][r] = p;
        rs[r] += p;
      }
#pragma unroll
    for (int off = 8; off >= 1; off >>= 1)
#pragma unroll
      for (int r = 0; r < 8; ++r) rs[r] += __shfl_xor(rs[r], off, 32);
#pragma unroll
    for (int r = 0; r < 8; ++r) {
      lrun[r] = lrun[r] * alpha[r] + rs[r];
      o0[r] *= alpha[r];
      o1[r] *= alpha[r];
    }

    // stage P (C-layout regs) -> LDS -> A-layout fragments
#pragma unroll
    for (int nt = 0; nt < 8; ++nt)
#pragma unroll
      for (int r = 0; r < 8; ++r)
        Ps[wave][r + 8 * half][nt * 16 + l16] = (__bf16)s[nt][r];
    asm volatile("s_wait_dscnt 0" ::: "memory");

    // O += P * V : K-dim 128 in four 32-chunks, two 16-wide d-tiles
#pragma unroll
    for (int kc = 0; kc < 4; ++kc) {
      v16bf pf = ld_frag(&Ps[wave][l16][kc * 32 + half * 8],
                         &Ps[wave][l16][kc * 32 + 16 + half * 8]);
      v16bf vf0 = ld_frag(&Vs[l16][kc * 32 + half * 8],
                          &Vs[l16][kc * 32 + 16 + half * 8]);
      v16bf vf1 = ld_frag(&Vs[16 + l16][kc * 32 + half * 8],
                          &Vs[16 + l16][kc * 32 + 16 + half * 8]);
      o0 = wmma_bf16(pf, vf0, o0);
      o1 = wmma_bf16(pf, vf1, o1);
    }
  }

#pragma unroll
  for (int r = 0; r < 8; ++r) {
    const float inv = 1.0f / lrun[r];
    const int trow = qrow0 + r;   // token index within batch b
    float* dst = O + ((size_t)b * TSEQ + trow) * DMODEL + hh * HDIM;
    dst[l16]      = o0[r] * inv;
    dst[16 + l16] = o1[r] * inv;
  }
}

// ---------------------------------------------------------------------------
// Minkowski LayerNorm: LN over dims [0,32) and [32,128) separately, * g + b.
// One wave per row (T_DIM = 32 == wave32).
// ---------------------------------------------------------------------------
__launch_bounds__(256)
__global__ void mln_kernel(const float* __restrict__ X, const float* __restrict__ g,
                           const float* __restrict__ be, float* __restrict__ Y) {
  const int lane = threadIdx.x & 31;
  const int wave = threadIdx.x >> 5;
  const size_t row = (size_t)blockIdx.x * 8 + wave;
  const float* x = X + row * DMODEL;
  float t0 = x[lane];
  float s1 = x[32 + lane], s2 = x[64 + lane], s3 = x[96 + lane];

  float mt = t0;
#pragma unroll
  for (int off = 16; off >= 1; off >>= 1) mt += __shfl_xor(mt, off, 32);
  mt *= (1.0f / 32.0f);
  const float d0 = t0 - mt;
  float vt = d0 * d0;
#pragma unroll
  for (int off = 16; off >= 1; off >>= 1) vt += __shfl_xor(vt, off, 32);
  vt *= (1.0f / 32.0f);
  const float rt = rsqrtf(vt + 1e-5f);

  float ms = s1 + s2 + s3;
#pragma unroll
  for (int off = 16; off >= 1; off >>= 1) ms += __shfl_xor(ms, off, 32);
  ms *= (1.0f / 96.0f);
  const float e1 = s1 - ms, e2 = s2 - ms, e3 = s3 - ms;
  float vs = e1 * e1 + e2 * e2 + e3 * e3;
#pragma unroll
  for (int off = 16; off >= 1; off >>= 1) vs += __shfl_xor(vs, off, 32);
  vs *= (1.0f / 96.0f);
  const float rsq = rsqrtf(vs + 1e-5f);

  float* y = Y + row * DMODEL;
  y[lane]      = d0 * rt  * g[lane]      + be[lane];
  y[32 + lane] = e1 * rsq * g[32 + lane] + be[32 + lane];
  y[64 + lane] = e2 * rsq * g[64 + lane] + be[64 + lane];
  y[96 + lane] = e3 * rsq * g[96 + lane] + be[96 + lane];
}

// ---------------------------------------------------------------------------
// Mean-pool over T then 128x2 classifier. One block per batch element.
// ---------------------------------------------------------------------------
__launch_bounds__(128)
__global__ void head_kernel(const float* __restrict__ H, const float* __restrict__ clsW,
                            const float* __restrict__ clsb, float* __restrict__ out) {
  __shared__ float pooled[DMODEL];
  const int b = blockIdx.x;
  const int d = threadIdx.x;
  float s = 0.0f;
  for (int t = 0; t < TSEQ; ++t) s += H[((size_t)b * TSEQ + t) * DMODEL + d];
  pooled[d] = s * (1.0f / (float)TSEQ);
  __syncthreads();
  if (d < 2) {
    float acc = clsb[d];
    for (int k = 0; k < DMODEL; ++k) acc += pooled[k] * clsW[k * 2 + d];
    out[b * 2 + d] = acc;
  }
}

// ---------------------------------------------------------------------------
extern "C" void kernel_launch(void* const* d_in, const int* in_sizes, int n_in,
                              void* d_out, int out_size, void* d_ws, size_t ws_size,
                              hipStream_t stream) {
  (void)in_sizes; (void)n_in; (void)out_size; (void)ws_size;
  const float* x       = (const float*)d_in[0];
  const float* embed_W = (const float*)d_in[1];
  const float* embed_b = (const float*)d_in[2];
  const float* Wq  = (const float*)d_in[3];
  const float* bq  = (const float*)d_in[4];
  const float* Wk  = (const float*)d_in[5];
  const float* bk  = (const float*)d_in[6];
  const float* Wv  = (const float*)d_in[7];
  const float* bv  = (const float*)d_in[8];
  const float* Wo  = (const float*)d_in[9];
  const float* bo  = (const float*)d_in[10];
  const float* g1  = (const float*)d_in[11];
  const float* b1  = (const float*)d_in[12];
  const float* W1  = (const float*)d_in[13];
  const float* bf1 = (const float*)d_in[14];
  const float* W2  = (const float*)d_in[15];
  const float* bf2 = (const float*)d_in[16];
  const float* g2  = (const float*)d_in[17];
  const float* b2  = (const float*)d_in[18];
  const float* clsW = (const float*)d_in[19];
  const float* clsb = (const float*)d_in[20];

  float* ws = (float*)d_ws;
  const size_t NT = (size_t)NTOK;
  float* h    = ws;
  float* q    = h    + NT * DMODEL;
  float* kbuf = q    + NT * DMODEL;
  float* vbuf = kbuf + NT * DMODEL;
  float* a    = vbuf + NT * DMODEL;
  float* tmp  = a    + NT * DMODEL;
  float* ff   = tmp  + NT * DMODEL;   // NT * DFF floats

  embed_kernel<<<dim3((NTOK * DMODEL) / 256), 256, 0, stream>>>(x, embed_W, embed_b, h);

  for (int i = 0; i < 3; ++i) {
    const size_t wOff = (size_t)i * DMODEL * DMODEL;
    const size_t bOff = (size_t)i * DMODEL;
    gemm_kernel<<<dim3(64, 2), 256, 0, stream>>>(h, Wq + wOff, bq + bOff, nullptr, q,    NTOK, DMODEL, DMODEL, 0, 1);
    gemm_kernel<<<dim3(64, 2), 256, 0, stream>>>(h, Wk + wOff, bk + bOff, nullptr, kbuf, NTOK, DMODEL, DMODEL, 0, 1);
    gemm_kernel<<<dim3(64, 2), 256, 0, stream>>>(h, Wv + wOff, bv + bOff, nullptr, vbuf, NTOK, DMODEL, DMODEL, 0, 1);
    attn_kernel<<<dim3(16, 16), 256, 0, stream>>>(q, kbuf, vbuf, a);
    gemm_kernel<<<dim3(64, 2), 256, 0, stream>>>(a, Wo + wOff, bo + bOff, h, tmp, NTOK, DMODEL, DMODEL, 0, 0);
    mln_kernel<<<dim3(NTOK / 8), 256, 0, stream>>>(tmp, g1 + bOff, b1 + bOff, h);
    gemm_kernel<<<dim3(64, 8), 256, 0, stream>>>(h, W1 + (size_t)i * DMODEL * DFF, bf1 + (size_t)i * DFF,
                                                 nullptr, ff, NTOK, DFF, DMODEL, 1, 0);
    gemm_kernel<<<dim3(64, 2), 256, 0, stream>>>(ff, W2 + (size_t)i * DFF * DMODEL, bf2 + bOff,
                                                 h, tmp, NTOK, DMODEL, DFF, 0, 0);
    mln_kernel<<<dim3(NTOK / 8), 256, 0, stream>>>(tmp, g2 + bOff, b2 + bOff, h);
  }

  head_kernel<<<dim3(BBATCH), 128, 0, stream>>>(h, clsW, clsb, (float*)d_out);
}
